// GraphAttentionHead_3135326126435
// MI455X (gfx1250) — compile-verified, matrix operand order checked by reading, not run on hardware
//
#include <hip/hip_runtime.h>
#include <math.h>

typedef __attribute__((ext_vector_type(16))) _Float16 v16h;
typedef __attribute__((ext_vector_type(8)))  float    v8f;

#define IN_F  256
#define OUT_F 128
#define GAT_ALPHA 0.2f

// k-offset within a 32-wide K tile for the 16-bit WMMA A/B lane layout:
// lanes 0-15: VGPR0..3 -> K=0..7, VGPR4..7 -> K=16..23; lanes 16-31: +8.
__device__ __forceinline__ int wmma_k_of(int laneHalf, int h) {
    int v = h >> 1, p = h & 1;
    int k = (v < 4) ? (2 * v + p) : (16 + 2 * (v - 4) + p);
    return k + (laneHalf ? 8 : 0);
}

// ---------------- Kernel 1: pack W (fp32 [256,128]) into fp16 WMMA-B layout ---------
// Flat layout: ((kt*8 + nt)*32 + lane)*16 + h   -> each lane's 16 halves contiguous.
__global__ void pack_W_kernel(const float* __restrict__ W, _Float16* __restrict__ Wp) {
    int idx = blockIdx.x * blockDim.x + threadIdx.x;
    if (idx >= (IN_F / 32) * (OUT_F / 16) * 32 * 16) return;
    int h    = idx & 15;
    int lane = (idx >> 4) & 31;
    int nt   = (idx >> 9) & 7;
    int kt   = idx >> 12;
    int laneHalf = lane >> 4;
    int n = nt * 16 + (lane & 15);
    int k = kt * 32 + wmma_k_of(laneHalf, h);
    Wp[idx] = (_Float16)W[(size_t)k * OUT_F + n];
}

// ---------------- Kernel 2: Wh = h @ W via v_wmma_f32_16x16x32_f16 ------------------
// W (64 KB fp16, shared by all 8 waves of the block) is staged into LDS once with
// gfx1250 async-to-LDS loads (ASYNCcnt). Per K tile: one ds_load clause for all 8 B
// fragments, one wait, then 8 independent WMMAs back-to-back to keep the XDL pipe fed.
__global__ void __launch_bounds__(256)
gemm_wmma_kernel(const float* __restrict__ hmat, const _Float16* __restrict__ Wp,
                 float* __restrict__ Wh, int N) {
    __shared__ __align__(128) _Float16 lWp[IN_F * OUT_F];   // 64 KB

    int lane = threadIdx.x & 31;
    int wave = threadIdx.x >> 5;

    // ---- async stage: 256 threads x 16 x b128 = 64 KB  (global -> LDS, no VGPRs) ----
    {
        unsigned ldsBase = (unsigned)(uintptr_t)(void*)&lWp[0]; // low 32 bits = LDS offset
        unsigned gOff    = (unsigned)threadIdx.x * 256u;
#pragma unroll
        for (int j = 0; j < 16; ++j) {
            unsigned lAddr = ldsBase + gOff + (unsigned)(j * 16);
            unsigned vOff  = gOff + (unsigned)(j * 16);
            asm volatile("global_load_async_to_lds_b128 %0, %1, %2"
                         :: "v"(lAddr), "v"(vOff), "s"(Wp) : "memory");
        }
        asm volatile("s_wait_asynccnt 0x0" ::: "memory");
    }
    __syncthreads();

    int mBase = (blockIdx.x * 8 + wave) * 16;
    if (mBase >= N) return;                      // wave-uniform: EXEC stays all-1s
    int laneHalf = lane >> 4;
    int m   = mBase + (lane & 15);
    int mld = (m < N) ? m : (N - 1);             // clamp tail loads (stores guarded)
    const float* hrow = hmat + (size_t)mld * IN_F;

    const v8f vzero = {0.f, 0.f, 0.f, 0.f, 0.f, 0.f, 0.f, 0.f};
    v8f acc[8];
#pragma unroll
    for (int nt = 0; nt < 8; ++nt) acc[nt] = vzero;

#pragma unroll
    for (int kt = 0; kt < 8; ++kt) {
        // A tile: halves 0..7 <- k0..k0+7 ; halves 8..15 <- k0+16..k0+23
        const float* p = hrow + kt * 32 + 8 * laneHalf;
        v16h a;
#pragma unroll
        for (int j = 0; j < 8; ++j) a[j]     = (_Float16)p[j];
#pragma unroll
        for (int j = 0; j < 8; ++j) a[8 + j] = (_Float16)p[16 + j];
        // Batch all 8 B fragments for this K tile, then issue 8 independent WMMAs.
        v16h b[8];
#pragma unroll
        for (int nt = 0; nt < 8; ++nt)
            b[nt] = *(const v16h*)(lWp + (((size_t)kt * 8 + nt) * 32 + lane) * 16);
#pragma unroll
        for (int nt = 0; nt < 8; ++nt)
            acc[nt] = __builtin_amdgcn_wmma_f32_16x16x32_f16(
                false, a, false, b[nt], (short)0, acc[nt], false, false);
    }
    // C layout: VGPR r, lane l -> m = r + 8*(l>=16), n = l&15
#pragma unroll
    for (int nt = 0; nt < 8; ++nt) {
#pragma unroll
        for (int r = 0; r < 8; ++r) {
            int mr = mBase + r + 8 * laneHalf;
            if (mr < N) Wh[(size_t)mr * OUT_F + nt * 16 + (lane & 15)] = acc[nt][r];
        }
    }
}

// ---------------- Kernel 3: f1 = Wh@a_src, f2 = Wh@a_dest (wave per node) -----------
__global__ void __launch_bounds__(256)
f1f2_kernel(const float* __restrict__ Wh, const float* __restrict__ a_src,
            const float* __restrict__ a_dest, float* __restrict__ f1,
            float* __restrict__ f2, int N) {
    int lane = threadIdx.x & 31;
    int wave = threadIdx.x >> 5;
    int i = blockIdx.x * 8 + wave;
    if (i >= N) return;
    const float* wrow = Wh + (size_t)i * OUT_F + lane * 4;
    float s1 = 0.f, s2 = 0.f;
#pragma unroll
    for (int j = 0; j < 4; ++j) {
        float w = wrow[j];
        s1 = fmaf(w, a_src[lane * 4 + j], s1);
        s2 = fmaf(w, a_dest[lane * 4 + j], s2);
    }
#pragma unroll
    for (int off = 16; off >= 1; off >>= 1) {
        s1 += __shfl_xor(s1, off, 32);
        s2 += __shfl_xor(s2, off, 32);
    }
    if (lane == 0) { f1[i] = s1; f2[i] = s2; }
}

// ---------------- Kernel 4: row_ptr via binary search over sorted row[] -------------
__global__ void rowptr_kernel(const int* __restrict__ row, int* __restrict__ row_ptr,
                              int N, int E) {
    int i = blockIdx.x * blockDim.x + threadIdx.x;
    if (i > N) return;
    int lo = 0, hi = E;
    while (lo < hi) {
        int mid = (lo + hi) >> 1;
        if (row[mid] < i) lo = mid + 1; else hi = mid;
    }
    row_ptr[i] = lo;
}

// ---------------- Kernel 5: segment softmax + SpMM + ELU (wave per row) -------------
__global__ void __launch_bounds__(256)
attn_spmm_kernel(const float* __restrict__ Wh, const float* __restrict__ f1,
                 const float* __restrict__ f2, const int* __restrict__ col,
                 const int* __restrict__ row_ptr, float* __restrict__ out, int N) {
    int lane = threadIdx.x & 31;
    int wave = threadIdx.x >> 5;
    int i = blockIdx.x * 8 + wave;
    if (i >= N) return;
    int start = row_ptr[i], end = row_ptr[i + 1];
    float* orow = out + (size_t)i * OUT_F + lane * 4;
    if (start >= end) {                         // empty segment -> elu(0) = 0
#pragma unroll
        for (int j = 0; j < 4; ++j) orow[j] = 0.f;
        return;
    }
    float f1i = f1[i];
    // pass 1: segment max of leakyrelu logits
    float mmax = -INFINITY;
    for (int e = start + lane; e < end; e += 32) {
        float s = f1i + f2[col[e]];
        s = (s > 0.f) ? s : GAT_ALPHA * s;
        mmax = fmaxf(mmax, s);
    }
#pragma unroll
    for (int off = 16; off >= 1; off >>= 1) mmax = fmaxf(mmax, __shfl_xor(mmax, off, 32));
    // pass 2: denom
    float denom = 0.f;
    for (int e = start + lane; e < end; e += 32) {
        float s = f1i + f2[col[e]];
        s = (s > 0.f) ? s : GAT_ALPHA * s;
        denom += __expf(s - mmax);
    }
#pragma unroll
    for (int off = 16; off >= 1; off >>= 1) denom += __shfl_xor(denom, off, 32);
    float inv = 1.f / denom;
    // pass 3: all 32 lanes cooperate per edge; lane owns feats 4*lane..4*lane+3
    float a0 = 0.f, a1 = 0.f, a2 = 0.f, a3 = 0.f;
    for (int e = start; e < end; ++e) {
        int c = col[e];
        float s = f1i + f2[c];
        s = (s > 0.f) ? s : GAT_ALPHA * s;
        float att = __expf(s - mmax) * inv;
        const float4 w = *(const float4*)(Wh + (size_t)c * OUT_F + lane * 4);
        a0 = fmaf(att, w.x, a0); a1 = fmaf(att, w.y, a1);
        a2 = fmaf(att, w.z, a2); a3 = fmaf(att, w.w, a3);
    }
    orow[0] = (a0 > 0.f) ? a0 : __expf(a0) - 1.f;
    orow[1] = (a1 > 0.f) ? a1 : __expf(a1) - 1.f;
    orow[2] = (a2 > 0.f) ? a2 : __expf(a2) - 1.f;
    orow[3] = (a3 > 0.f) ? a3 : __expf(a3) - 1.f;
}

extern "C" void kernel_launch(void* const* d_in, const int* in_sizes, int n_in,
                              void* d_out, int out_size, void* d_ws, size_t ws_size,
                              hipStream_t stream) {
    const float* h      = (const float*)d_in[0];
    const float* W      = (const float*)d_in[1];
    const float* a_src  = (const float*)d_in[2];
    const float* a_dest = (const float*)d_in[3];
    const int*   row    = (const int*)d_in[4];
    const int*   col    = (const int*)d_in[5];
    int N = in_sizes[0] / IN_F;
    int E = in_sizes[4];

    char* ws = (char*)d_ws;
    size_t off = 0;
    _Float16* Wp = (_Float16*)(ws + off); off += (size_t)IN_F * OUT_F * sizeof(_Float16);
    off = (off + 255) & ~(size_t)255;
    float* Wh = (float*)(ws + off);       off += (size_t)N * OUT_F * sizeof(float);
    off = (off + 255) & ~(size_t)255;
    float* f1 = (float*)(ws + off);       off += (size_t)N * sizeof(float);
    off = (off + 255) & ~(size_t)255;
    float* f2 = (float*)(ws + off);       off += (size_t)N * sizeof(float);
    off = (off + 255) & ~(size_t)255;
    int* row_ptr = (int*)(ws + off);

    pack_W_kernel<<<(IN_F * OUT_F + 255) / 256, 256, 0, stream>>>(W, Wp);
    gemm_wmma_kernel<<<(N + 127) / 128, 256, 0, stream>>>(h, Wp, Wh, N);
    f1f2_kernel<<<(N + 7) / 8, 256, 0, stream>>>(Wh, a_src, a_dest, f1, f2, N);
    rowptr_kernel<<<(N + 1 + 255) / 256, 256, 0, stream>>>(row, row_ptr, N, E);
    attn_spmm_kernel<<<(N + 7) / 8, 256, 0, stream>>>(Wh, f1, f2, col, row_ptr,
                                                      (float*)d_out, N);
}
